// AttentionLSTMDecoder_36275293782195
// MI455X (gfx1250) — matrix operand order repaired and checked by reading, not compiled, not run on hardware
//
#include <hip/hip_runtime.h>
#include <hip/hip_bf16.h>

// Problem constants (match reference)
constexpr int Vv = 30000, Ee = 256, Hh = 512, Ff = 2048;
constexpr int Bb = 64, Ss = 20, HW = 49, Gg = 4 * Hh;

typedef __bf16 bf16_t;
typedef bf16_t v16bf __attribute__((ext_vector_type(16)));
typedef bf16_t v8bf  __attribute__((ext_vector_type(8)));
typedef float  v8f   __attribute__((ext_vector_type(8)));

static __device__ __forceinline__ bf16_t f2bf(float x) {
  unsigned u = __builtin_bit_cast(unsigned, x);
  unsigned r = u + 0x7FFFu + ((u >> 16) & 1u);   // round-to-nearest-even
  unsigned short h = (unsigned short)(r >> 16);
  return __builtin_bit_cast(bf16_t, h);
}

static __device__ __forceinline__ float sigm(float x) {
  return 1.0f / (1.0f + __expf(-x));
}

// ---------------- one-time prep kernels ----------------

__global__ void cvt_bf16_kernel(const float* __restrict__ src,
                                bf16_t* __restrict__ dst, int n) {
  int i = blockIdx.x * blockDim.x + threadIdx.x;
  if (i < n) dst[i] = f2bf(src[i]);
}

// src: (B, F, 49) f32  ->  dst: (B*49, F) bf16
__global__ void feat_transpose_kernel(const float* __restrict__ src,
                                      bf16_t* __restrict__ dst) {
  int i = blockIdx.x * blockDim.x + threadIdx.x;
  const int total = Bb * HW * Ff;
  if (i >= total) return;
  int f = i % Ff;
  int bp = i / Ff;
  int p = bp % HW;
  int b = bp / HW;
  dst[i] = f2bf(src[((size_t)b * Ff + f) * HW + p]);
}

// word_ids (B,S) int; table (V,E) f32 -> xs (S,B,E) bf16 with padding_idx=0
__global__ void embed_kernel(const int* __restrict__ ids,
                             const float* __restrict__ table,
                             bf16_t* __restrict__ xs) {
  int i = blockIdx.x * blockDim.x + threadIdx.x;
  const int total = Ss * Bb * Ee;
  if (i >= total) return;
  int e = i % Ee;
  int tb = i / Ee;
  int b = tb % Bb;
  int t = tb / Bb;
  int id = ids[b * Ss + t];
  xs[i] = (id != 0) ? f2bf(table[(size_t)id * Ee + e]) : f2bf(0.0f);
}

// ---- generic WMMA GEMM: C[M,N] (+)= A[M,K] * W[N,K]^T + bias ----
// A bf16 row-major (lda), W bf16 row-major N x K (ldw), C f32 (ldc).
// M multiple of 16*RT, N multiple of 16, K multiple of 32.
// One wave owns a (16*RT) x 16 column strip: the weight (B) fragment is
// loaded once per k-chunk and reused across RT independent WMMA chains.
template <int RT>
__global__ __launch_bounds__(256)
void wmma_gemm_kernel(const bf16_t* __restrict__ A, int lda,
                      const bf16_t* __restrict__ W, int ldw,
                      float* __restrict__ C, int ldc,
                      int M, int N, int K,
                      const float* __restrict__ bias, int accum) {
  const int lane = threadIdx.x & 31;
  const int wave = threadIdx.x >> 5;
  const int sel  = lane >> 4;     // 0: lanes 0-15, 1: lanes 16-31
  const int l16  = lane & 15;
  const int tilesN  = N >> 4;
  const int groupsM = M / (16 * RT);
  const int numGroups = groupsM * tilesN;
  const int stride = gridDim.x * 8;

  for (int g = blockIdx.x * 8 + wave; g < numGroups; g += stride) {
    const int m0 = (g / tilesN) * (16 * RT);
    const int n0 = (g % tilesN) << 4;
    // B fragment: lane l16 -> col n0+l16 ; 16 contiguous K at sel*16
    const bf16_t* Wrow = W + (size_t)(n0 + l16) * ldw + sel * 16;
    // A fragments: lane l16 -> row ; 16 contiguous K at sel*16
    const bf16_t* Arow[RT];
    v8f acc[RT];
#pragma unroll
    for (int r = 0; r < RT; ++r) {
      Arow[r] = A + (size_t)(m0 + 16 * r + l16) * lda + sel * 16;
      acc[r] = (v8f){0.f, 0.f, 0.f, 0.f, 0.f, 0.f, 0.f, 0.f};
    }

    for (int k0 = 0; k0 < K; k0 += 32) {
      v16bf b = *(const v16bf*)(Wrow + k0);
#pragma unroll
      for (int r = 0; r < RT; ++r) {
        v8bf alo = *(const v8bf*)(Arow[r] + k0);        // K = k0+sel*16 .. +7
        v8bf ahi = *(const v8bf*)(Arow[r] + k0 + 8);    // K = k0+sel*16+8 ..
        v16bf a = __builtin_shufflevector(alo, ahi, 0, 1, 2, 3, 4, 5, 6, 7,
                                          8, 9, 10, 11, 12, 13, 14, 15);
        acc[r] = __builtin_amdgcn_wmma_f32_16x16x32_bf16(
            false, a, false, b, (short)0, acc[r], false, false);
      }
    }

    const int n = n0 + l16;
    const float bv = bias ? bias[n] : 0.0f;
#pragma unroll
    for (int r = 0; r < RT; ++r) {
#pragma unroll
      for (int q = 0; q < 8; ++q) {
        const int m = m0 + 16 * r + q + sel * 8;  // C/D: VGPR q -> M = q + 8*sel
        float* cp = C + (size_t)m * ldc + n;
        float prev = accum ? *cp : 0.0f;
        *cp = acc[r][q] + prev + bv;
      }
    }
  }
}

// ---------------- per-step small kernels ----------------

// scores = ha . proj ; softmax over 49 ; ctx = w . proj  (one block per batch)
__global__ void attn_kernel(const float* __restrict__ ha,
                            const float* __restrict__ proj,
                            bf16_t* __restrict__ ctxb) {
  __shared__ float sw[HW];
  const int b = blockIdx.x;
  const int t = threadIdx.x;               // 0..511
  const float* pb = proj + (size_t)b * HW * Hh;
  const float* hb = ha + (size_t)b * Hh;
  if (t < HW) {
    float s = 0.0f;
    for (int k = 0; k < Hh; ++k) s += hb[k] * pb[t * Hh + k];
    sw[t] = s;
  }
  __syncthreads();
  if (t == 0) {
    float mx = sw[0];
    for (int p = 1; p < HW; ++p) mx = fmaxf(mx, sw[p]);
    float sum = 0.0f;
    for (int p = 0; p < HW; ++p) { float e = __expf(sw[p] - mx); sw[p] = e; sum += e; }
    float inv = 1.0f / sum;
    for (int p = 0; p < HW; ++p) sw[p] *= inv;
  }
  __syncthreads();
  float acc = 0.0f;
  for (int p = 0; p < HW; ++p) acc += sw[p] * pb[p * Hh + t];
  ctxb[(size_t)b * Hh + t] = f2bf(acc);
}

// gates (B,4H) f32, c (B,H) f32 in-place; writes h f32 + bf16
__global__ void lstm_elem_kernel(const float* __restrict__ gates,
                                 float* __restrict__ c,
                                 float* __restrict__ hf,
                                 bf16_t* __restrict__ hb) {
  int i = blockIdx.x * blockDim.x + threadIdx.x;
  if (i >= Bb * Hh) return;
  int b = i / Hh, j = i % Hh;
  const float* g = gates + (size_t)b * Gg;
  float gi = sigm(g[j]);
  float gf = sigm(g[Hh + j]);
  float gg = tanhf(g[2 * Hh + j]);
  float go = sigm(g[3 * Hh + j]);
  float cn = gf * c[i] + gi * gg;
  c[i] = cn;
  float hn = go * tanhf(cn);
  hf[i] = hn;
  hb[i] = f2bf(hn);
}

// v = tanh(src); replicate to two f32 and two bf16 destinations (layer0/layer1 init)
__global__ void tanh2_kernel(const float* __restrict__ src,
                             float* __restrict__ d0f, float* __restrict__ d1f,
                             bf16_t* __restrict__ d0b, bf16_t* __restrict__ d1b,
                             int n) {
  int i = blockIdx.x * blockDim.x + threadIdx.x;
  if (i >= n) return;
  float v = tanhf(src[i]);
  d0f[i] = v; d1f[i] = v;
  if (d0b) d0b[i] = f2bf(v);
  if (d1b) d1b[i] = f2bf(v);
}

// ---------------- host orchestration ----------------

extern "C" void kernel_launch(void* const* d_in, const int* in_sizes, int n_in,
                              void* d_out, int out_size, void* d_ws, size_t ws_size,
                              hipStream_t stream) {
  (void)in_sizes; (void)n_in; (void)out_size; (void)ws_size;

  const int*   word_ids = (const int*)d_in[0];
  const float* cnn      = (const float*)d_in[1];
  const float* attnF    = (const float*)d_in[2];
  const float* embTab   = (const float*)d_in[3];
  const float* Wf   = (const float*)d_in[4];  const float* bf_p = (const float*)d_in[5];
  const float* Wa   = (const float*)d_in[6];  const float* ba   = (const float*)d_in[7];
  const float* Wih0 = (const float*)d_in[8];  const float* Whh0 = (const float*)d_in[9];
  const float* bih0 = (const float*)d_in[10]; const float* bhh0 = (const float*)d_in[11];
  const float* Wih1 = (const float*)d_in[12]; const float* Whh1 = (const float*)d_in[13];
  const float* bih1 = (const float*)d_in[14]; const float* bhh1 = (const float*)d_in[15];
  const float* Whi  = (const float*)d_in[16]; const float* bhi  = (const float*)d_in[17];
  const float* Wci  = (const float*)d_in[18]; const float* bci  = (const float*)d_in[19];
  const float* Wout = (const float*)d_in[20]; const float* bout = (const float*)d_in[21];

  float* out = (float*)d_out;

  // ---- workspace carve-up (256B aligned) ----
  char* wp = (char*)d_ws;
  auto alloc = [&](size_t bytes) -> void* {
    void* p = (void*)wp;
    wp += (bytes + 255) & ~((size_t)255);
    return p;
  };
  bf16_t* Wf_b   = (bf16_t*)alloc((size_t)Hh * Ff * 2);
  bf16_t* Wa_b   = (bf16_t*)alloc((size_t)Hh * Hh * 2);
  bf16_t* Wih0_b = (bf16_t*)alloc((size_t)Gg * (Ee + Hh) * 2);
  bf16_t* Whh0_b = (bf16_t*)alloc((size_t)Gg * Hh * 2);
  bf16_t* Wih1_b = (bf16_t*)alloc((size_t)Gg * Hh * 2);
  bf16_t* Whh1_b = (bf16_t*)alloc((size_t)Gg * Hh * 2);
  bf16_t* Whi_b  = (bf16_t*)alloc((size_t)Hh * Hh * 2);
  bf16_t* Wci_b  = (bf16_t*)alloc((size_t)Hh * Hh * 2);
  bf16_t* Wout_b = (bf16_t*)alloc((size_t)Vv * Hh * 2);
  bf16_t* cnn_b  = (bf16_t*)alloc((size_t)Bb * Hh * 2);
  bf16_t* featb  = (bf16_t*)alloc((size_t)Bb * HW * Ff * 2);
  bf16_t* xsb    = (bf16_t*)alloc((size_t)Ss * Bb * Ee * 2);
  bf16_t* ctxb   = (bf16_t*)alloc((size_t)Bb * Hh * 2);
  bf16_t* h0b    = (bf16_t*)alloc((size_t)Bb * Hh * 2);
  bf16_t* h1b    = (bf16_t*)alloc((size_t)Bb * Hh * 2);
  float*  proj   = (float*)alloc((size_t)Bb * HW * Hh * 4);
  float*  ha     = (float*)alloc((size_t)Bb * Hh * 4);
  float*  gates  = (float*)alloc((size_t)Bb * Gg * 4);
  float*  tmp    = (float*)alloc((size_t)Bb * Hh * 4);
  float*  h0f    = (float*)alloc((size_t)Bb * Hh * 4);
  float*  h1f    = (float*)alloc((size_t)Bb * Hh * 4);
  float*  c0     = (float*)alloc((size_t)Bb * Hh * 4);
  float*  c1     = (float*)alloc((size_t)Bb * Hh * 4);

  auto cvt = [&](const float* s, bf16_t* d, int n) {
    cvt_bf16_kernel<<<(n + 255) / 256, 256, 0, stream>>>(s, d, n);
  };
  // M must be a multiple of 64 (RT=4) — true for all call sites (64 and 3136).
  auto gemm = [&](const bf16_t* A, int lda, const bf16_t* W, int ldw,
                  float* C, int ldc, int M, int N, int K,
                  const float* bias, int accum) {
    int groups = (M / 64) * (N / 16);
    int blocks = (groups + 7) / 8;
    wmma_gemm_kernel<4><<<blocks, 256, 0, stream>>>(A, lda, W, ldw, C, ldc,
                                                    M, N, K, bias, accum);
  };

  // ---- one-time prep ----
  cvt(Wf,   Wf_b,   Hh * Ff);
  cvt(Wa,   Wa_b,   Hh * Hh);
  cvt(Wih0, Wih0_b, Gg * (Ee + Hh));
  cvt(Whh0, Whh0_b, Gg * Hh);
  cvt(Wih1, Wih1_b, Gg * Hh);
  cvt(Whh1, Whh1_b, Gg * Hh);
  cvt(Whi,  Whi_b,  Hh * Hh);
  cvt(Wci,  Wci_b,  Hh * Hh);
  cvt(Wout, Wout_b, Vv * Hh);
  cvt(cnn,  cnn_b,  Bb * Hh);

  {
    int n = Bb * HW * Ff;
    feat_transpose_kernel<<<(n + 255) / 256, 256, 0, stream>>>(attnF, featb);
  }
  {
    int n = Ss * Bb * Ee;
    embed_kernel<<<(n + 255) / 256, 256, 0, stream>>>(word_ids, embTab, xsb);
  }

  // h_init / c_init = tanh(cnn @ W^T + b), replicated to both layers
  gemm(cnn_b, Hh, Whi_b, Hh, tmp, Hh, Bb, Hh, Hh, bhi, 0);
  tanh2_kernel<<<(Bb * Hh + 255) / 256, 256, 0, stream>>>(tmp, h0f, h1f, h0b, h1b, Bb * Hh);
  gemm(cnn_b, Hh, Wci_b, Hh, tmp, Hh, Bb, Hh, Hh, bci, 0);
  tanh2_kernel<<<(Bb * Hh + 255) / 256, 256, 0, stream>>>(tmp, c0, c1, (bf16_t*)nullptr, (bf16_t*)nullptr, Bb * Hh);

  // proj = feat @ Wf^T + bf : (3136, 512)
  gemm(featb, Ff, Wf_b, Ff, proj, Hh, Bb * HW, Hh, Ff, bf_p, 0);

  // ---- decode loop (sequenced on stream) ----
  for (int t = 0; t < Ss; ++t) {
    // ha = h1 @ Wa^T + ba
    gemm(h1b, Hh, Wa_b, Hh, ha, Hh, Bb, Hh, Hh, ba, 0);
    // attention -> ctx (bf16)
    attn_kernel<<<Bb, Hh, 0, stream>>>(ha, proj, ctxb);

    // layer-0 gates = [x_t, ctx] @ Wih0^T + h0 @ Whh0^T + biases
    gemm(xsb + (size_t)t * Bb * Ee, Ee, Wih0_b, Ee + Hh,
         gates, Gg, Bb, Gg, Ee, bih0, 0);
    gemm(ctxb, Hh, Wih0_b + Ee, Ee + Hh, gates, Gg, Bb, Gg, Hh, bhh0, 1);
    gemm(h0b, Hh, Whh0_b, Hh, gates, Gg, Bb, Gg, Hh, (const float*)nullptr, 1);
    lstm_elem_kernel<<<(Bb * Hh + 255) / 256, 256, 0, stream>>>(gates, c0, h0f, h0b);

    // layer-1 gates = h0_new @ Wih1^T + h1_old @ Whh1^T + biases
    gemm(h0b, Hh, Wih1_b, Hh, gates, Gg, Bb, Gg, Hh, bih1, 0);
    gemm(h1b, Hh, Whh1_b, Hh, gates, Gg, Bb, Gg, Hh, bhh1, 1);
    lstm_elem_kernel<<<(Bb * Hh + 255) / 256, 256, 0, stream>>>(gates, c1, h1f, h1b);

    // logits: write directly into d_out[b, t, :] (row stride S*V)
    gemm(h1b, Hh, Wout_b, Hh, out + (size_t)t * Vv, Ss * Vv,
         Bb, Vv, Hh, bout, 0);
  }

  // ---- final h/c: d_out tail = h0, h1, c0, c1 (each B*H) ----
  float* tail = out + (size_t)Bb * Ss * Vv;
  size_t bytes = (size_t)Bb * Hh * sizeof(float);
  hipMemcpyAsync(tail,               h0f, bytes, hipMemcpyDeviceToDevice, stream);
  hipMemcpyAsync(tail + Bb * Hh,     h1f, bytes, hipMemcpyDeviceToDevice, stream);
  hipMemcpyAsync(tail + 2 * Bb * Hh, c0,  bytes, hipMemcpyDeviceToDevice, stream);
  hipMemcpyAsync(tail + 3 * Bb * Hh, c1,  bytes, hipMemcpyDeviceToDevice, stream);
}
